// DiscriminatorWithFilterConditioning_81930796138633
// MI455X (gfx1250) — compile-verified, matrix operand order checked by reading, not run on hardware
//
#include <hip/hip_runtime.h>
#include <math.h>

// ---------------------------------------------------------------------------
// DiscriminatorWithFilterConditioning for MI455X (gfx1250, wave32, WMMA)
// bf16 NHWC activations end-to-end; weights pre-transposed/converted to bf16;
// channel mixers = double-buffered LDS-tiled bf16 WMMA GEMMs with fused
// bias + LeakyReLU + BN-stat epilogue (sigmoid on the final layer).
// ---------------------------------------------------------------------------

typedef __attribute__((ext_vector_type(16))) __bf16 bf16x16;
typedef __attribute__((ext_vector_type(8)))  __bf16 bf16x8;
typedef __attribute__((ext_vector_type(8)))  float  f32x8;

__device__ __forceinline__ __bf16 f2bf(float f) {
  unsigned u = __builtin_bit_cast(unsigned, f);
  unsigned r = (u + 0x7FFFu + ((u >> 16) & 1u)) >> 16;   // round-nearest-even
  return __builtin_bit_cast(__bf16, (unsigned short)r);
}
__device__ __forceinline__ float bf2f(__bf16 h) {
  unsigned u = (unsigned)__builtin_bit_cast(unsigned short, h) << 16;
  return __builtin_bit_cast(float, u);
}
__device__ __forceinline__ __bf16 bfzero() {
  return __builtin_bit_cast(__bf16, (unsigned short)0);
}

// ------- weight prep: W (K,D) f32 -> Wt (D,Kpad) bf16, zero-padded ---------
__global__ void wprep_kernel(const float* __restrict__ W,
                             __bf16* __restrict__ Wt,
                             int K, int Kpad, int D, int total) {
  int idx = blockIdx.x * blockDim.x + threadIdx.x;
  if (idx >= total) return;
  int d = idx / Kpad, kp = idx - d * Kpad;
  float v = (kp < K) ? W[(size_t)kp * D + d] : 0.f;
  Wt[idx] = f2bf(v);
}

// ------- dynamic depthwise kernels: tanh(emb[label] @ W + b), (b,khw,c) -----
__global__ void dynk_kernel(const int* __restrict__ label,
                            const float* __restrict__ emb,
                            const float* __restrict__ w,
                            const float* __restrict__ bias,
                            float* __restrict__ out, int C, int ksq,
                            int total) {
  int idx = blockIdx.x * blockDim.x + threadIdx.x;
  if (idx >= total) return;
  const int N = C * ksq;
  int b = idx / N, o = idx - b * N;
  int khw = o / C, c = o - khw * C;
  int j = c * ksq + khw;                       // source column in (EMB, C*k*k)
  const float* e = emb + label[b] * 5;
  float acc = bias[j];
#pragma unroll
  for (int i = 0; i < 5; ++i) acc += e[i] * w[i * N + j];
  out[idx] = tanhf(acc);
}

// ------- layer-1 depthwise conv: f32 NCHW in -> bf16 NHWC out (8ch pad) -----
__global__ void dwconv1_kernel(const float* __restrict__ X,   // (128,3,128,128)
                               const float* __restrict__ Kn,  // (128,16,3)
                               __bf16* __restrict__ Y,        // (128,64*64,8)
                               int total) {
  int idx = blockIdx.x * blockDim.x + threadIdx.x;
  if (idx >= total) return;
  int c  = idx & 7;  int t = idx >> 3;
  int wo = t & 63;   t >>= 6;
  int ho = t & 63;   int b = t >> 6;
  float acc = 0.f;
  if (c < 3) {
    const float* xp = X + ((size_t)b * 3 + c) * 16384;
    const float* kp = Kn + (size_t)b * 48 + c;
    int hi0 = ho * 2 - 1, wi0 = wo * 2 - 1;
#pragma unroll
    for (int kh = 0; kh < 4; ++kh) {
      int hi = hi0 + kh;
      if ((unsigned)hi >= 128u) continue;
#pragma unroll
      for (int kw = 0; kw < 4; ++kw) {
        int wi = wi0 + kw;
        if ((unsigned)wi >= 128u) continue;
        acc += xp[hi * 128 + wi] * kp[(kh * 4 + kw) * 3];
      }
    }
  }
  Y[idx] = f2bf(acc);
}

// ------- generic depthwise conv, bf16 NHWC, 8 channels per thread ----------
__global__ void dwconv_nhwc_kernel(const __bf16* __restrict__ X,
                                   const float* __restrict__ Kn, // (b,khw,c)
                                   __bf16* __restrict__ Y,
                                   int C, int Hin, int Hout, int ks,
                                   int stride, int pad, int total8) {
  int idx = blockIdx.x * blockDim.x + threadIdx.x;
  if (idx >= total8) return;
  const size_t base = (size_t)idx * 8;
  int c0 = (int)(base % (size_t)C);
  long px = (long)(base / (size_t)C);
  int wo = (int)(px % Hout); px /= Hout;
  int ho = (int)(px % Hout); int b = (int)(px / Hout);
  const __bf16* xp = X + (size_t)b * Hin * Hin * C + c0;
  const float*  kp = Kn + (size_t)b * ks * ks * C + c0;
  int hi0 = ho * stride - pad, wi0 = wo * stride - pad;
  f32x8 acc = {0.f, 0.f, 0.f, 0.f, 0.f, 0.f, 0.f, 0.f};
  for (int kh = 0; kh < ks; ++kh) {
    int hi = hi0 + kh;
    if ((unsigned)hi >= (unsigned)Hin) continue;
    for (int kw = 0; kw < ks; ++kw) {
      int wi = wi0 + kw;
      if ((unsigned)wi >= (unsigned)Hin) continue;
      bf16x8 xv = *(const bf16x8*)(xp + (size_t)(hi * Hin + wi) * C);
      const float4* kf = (const float4*)(kp + (size_t)(kh * ks + kw) * C);
      float4 k0 = kf[0], k1 = kf[1];
      acc[0] += bf2f(xv[0]) * k0.x;
      acc[1] += bf2f(xv[1]) * k0.y;
      acc[2] += bf2f(xv[2]) * k0.z;
      acc[3] += bf2f(xv[3]) * k0.w;
      acc[4] += bf2f(xv[4]) * k1.x;
      acc[5] += bf2f(xv[5]) * k1.y;
      acc[6] += bf2f(xv[6]) * k1.z;
      acc[7] += bf2f(xv[7]) * k1.w;
    }
  }
  bf16x8 o;
#pragma unroll
  for (int i = 0; i < 8; ++i) o[i] = f2bf(acc[i]);
  *(bf16x8*)(Y + base) = o;
}

// ---------------- fused channel mix: Y = act(X Wt^T + bias), + BN stats -----
// X:  (Btot, KX)  bf16 row-major (NHWC).
// Wt: (D, Kpad)   bf16 row-major (pre-transposed, zero-padded).
// Block = 256 threads = 8 waves, 64(M) x 64(N) tile; wave owns 1 N sub-tile
// and 2 M sub-tiles (shared B fragment, 2 WMMAs/slab). K-slab = 32,
// double-buffered LDS. Both slab fills: 1x b128 load + 1x b128 LDS store
// per thread, no conversion. EXACT: KX == Kpad (layers 2..5).
// FINAL: sigmoid epilogue, f32 NCHW output; else bf16x8 NHWC + BN stats.
#define MT 64
#define NT 64
#define KT 32
#define LROW 40   // padded LDS row stride in bf16 elements (multiple of 8)

template <bool EXACT, bool FINAL>
__global__ __launch_bounds__(256)
void mix_gemm_kernel(const __bf16* __restrict__ X,   // (Btot, KX) bf16
                     const __bf16* __restrict__ Wt,  // (D, Kpad) bf16
                     const float* __restrict__ bias, // (D)
                     void* __restrict__ Yout,
                     float* __restrict__ s1,         // (D) sum
                     float* __restrict__ s2,         // (D) sumsq
                     int KX, int Kpad, int D, int HW, int Btot) {
  __shared__ __align__(16) __bf16 As[2][MT * LROW];
  __shared__ __align__(16) __bf16 Bs[2][NT * LROW];

  const int tid  = threadIdx.x;
  const int lane = tid & 31;
  const int wave = tid >> 5;
  const int nblk = Btot / NT;
  const int mb   = blockIdx.x / nblk;
  const int nb   = blockIdx.x - mb * nblk;
  const int m0   = mb * MT;
  const int n0   = nb * NT;

  const int mw = wave >> 2;              // 0..1
  const int nw = wave & 3;               // 0..3
  const int l15  = lane & 15;
  const int half = lane >> 4;            // K-half per ISA 16-bit layout
  const int kbase = half * 8;

  const int mA = mw * 16 + l15;          // first A fragment row in block tile
  const int nB = nw * 16 + l15;          // B fragment col in block tile
  const int n  = n0 + nB;                // flattened B*HW pixel index

  // fill coords (same for A and B): 64 rows x 32 k, 8 bf16 per thread
  const int rr = tid & 63;               // row within tile (m or n)
  const int kc = (tid >> 6) << 3;        // 0,8,16,24
  const __bf16* aTB = Wt + (size_t)(m0 + rr) * Kpad + kc;
  const __bf16* bTB = X + (size_t)(n0 + rr) * KX + kc;

  const int nslab = Kpad / KT;

  auto fill = [&](int s, int buf) {
    const int k0 = s * KT;
    // A slab: pure bf16 copy (Wt is zero-padded to Kpad)
    *(bf16x8*)(&As[buf][rr * LROW + kc]) = *(const bf16x8*)(aTB + k0);
    // B slab: pure bf16 copy; layer-1 guard for KX < Kpad
    __bf16 z = bfzero();
    bf16x8 xv = {z, z, z, z, z, z, z, z};
    if (EXACT || (k0 + kc) < KX) xv = *(const bf16x8*)(bTB + k0);
    *(bf16x8*)(&Bs[buf][rr * LROW + kc]) = xv;
  };

  f32x8 acc0 = {0.f, 0.f, 0.f, 0.f, 0.f, 0.f, 0.f, 0.f};
  f32x8 acc1 = acc0;

  fill(0, 0);
  for (int s = 0; s < nslab; ++s) {
    __syncthreads();                     // slab s resident; buf (s+1)&1 free
    if (s + 1 < nslab) fill(s + 1, (s + 1) & 1);

    const __bf16* ab = As[s & 1];
    const __bf16* bb = Bs[s & 1];
    const bf16x8* ap0 = (const bf16x8*)(ab + mA * LROW + kbase);
    const bf16x8* ap1 = (const bf16x8*)(ab + (mA + 32) * LROW + kbase);
    const bf16x8* bp  = (const bf16x8*)(bb + nB * LROW + kbase);
    bf16x16 af0 = __builtin_shufflevector(ap0[0], ap0[2], 0, 1, 2, 3, 4, 5, 6,
                                          7, 8, 9, 10, 11, 12, 13, 14, 15);
    bf16x16 af1 = __builtin_shufflevector(ap1[0], ap1[2], 0, 1, 2, 3, 4, 5, 6,
                                          7, 8, 9, 10, 11, 12, 13, 14, 15);
    bf16x16 bf  = __builtin_shufflevector(bp[0], bp[2], 0, 1, 2, 3, 4, 5, 6,
                                          7, 8, 9, 10, 11, 12, 13, 14, 15);
    acc0 = __builtin_amdgcn_wmma_f32_16x16x32_bf16(false, af0, false, bf,
                                                   (short)0, acc0, false, false);
    acc1 = __builtin_amdgcn_wmma_f32_16x16x32_bf16(false, af1, false, bf,
                                                   (short)0, acc1, false, false);
  }

  // ---- epilogue: D layout = VGPR r -> (M = mo0 + r, N = l15) ----
#pragma unroll
  for (int j = 0; j < 2; ++j) {
    const f32x8& acc = j ? acc1 : acc0;
    const int mo0 = m0 + mw * 16 + j * 32 + half * 8;   // 8 consecutive chans
    const float4* bp4 = (const float4*)(bias + mo0);
    float4 bA = bp4[0], bB = bp4[1];
    const float bv[8] = {bA.x, bA.y, bA.z, bA.w, bB.x, bB.y, bB.z, bB.w};
    if (FINAL) {
      float* Yf = (float*)Yout;                         // f32 NCHW to d_out
      const int bq = n / HW, hq = n - bq * HW;
#pragma unroll
      for (int r = 0; r < 8; ++r) {
        float v = acc[r] + bv[r];
        v = 1.f / (1.f + expf(-v));
        Yf[((size_t)bq * D + (mo0 + r)) * HW + hq] = v;
      }
    } else {
      __bf16* Yh = (__bf16*)Yout;                       // bf16 NHWC
      bf16x8 o;
#pragma unroll
      for (int r = 0; r < 8; ++r) {
        float v = acc[r] + bv[r];
        v = (v > 0.f) ? v : 0.2f * v;
        o[r] = f2bf(v);
        float t1 = v, t2 = v * v;
        for (int off = 8; off; off >>= 1) {   // reduce 16-lane N group
          t1 += __shfl_xor(t1, off, 16);
          t2 += __shfl_xor(t2, off, 16);
        }
        if (l15 == 0) {
          atomicAdd(&s1[mo0 + r], t1);
          atomicAdd(&s2[mo0 + r], t2);
        }
      }
      *(bf16x8*)(Yh + (size_t)n * D + mo0) = o;         // one 16B store
    }
  }
}

// ---------------- BN finalize on bf16 NHWC, 8 channels per thread -----------
__global__ void bn_bf16_kernel(__bf16* __restrict__ Y,
                               const float* __restrict__ s1,
                               const float* __restrict__ s2,
                               const float* __restrict__ g,
                               const float* __restrict__ bta,
                               int D, int total8, float invN) {
  int idx = blockIdx.x * blockDim.x + threadIdx.x;
  if (idx >= total8) return;
  const size_t base = (size_t)idx * 8;
  const int d0 = (int)(base % (size_t)D);
  bf16x8 v = *(bf16x8*)(Y + base);
  bf16x8 o;
#pragma unroll
  for (int i = 0; i < 8; ++i) {
    const int d = d0 + i;
    float mean = s1[d] * invN;
    float var  = s2[d] * invN - mean * mean;
    float sc   = rsqrtf(var + 1e-5f) * g[d];
    o[i] = f2bf((bf2f(v[i]) - mean) * sc + bta[d]);
  }
  *(bf16x8*)(Y + base) = o;
}

__global__ void zero_kernel(float* __restrict__ p, int n) {
  int idx = blockIdx.x * blockDim.x + threadIdx.x;
  if (idx < n) p[idx] = 0.f;
}

__global__ void label_copy_kernel(const int* __restrict__ lab,
                                  float* __restrict__ out, int n) {
  int idx = blockIdx.x * blockDim.x + threadIdx.x;
  if (idx < n) out[idx] = (float)lab[idx];
}

// ---------------------------------------------------------------------------
static inline int ceil_div(long a, long b) { return (int)((a + b - 1) / b); }

static void launch_mix(const __bf16* X, const __bf16* Wt, const float* bias,
                       void* Y, float* s1, float* s2,
                       int KX, int Kpad, int D, int HW, int final_layer,
                       hipStream_t stream) {
  const int Btot = 128 * HW;
  const int blocks = (D / MT) * (Btot / NT);   // exact for every layer
  if (final_layer)
    mix_gemm_kernel<true, true><<<blocks, 256, 0, stream>>>(
        X, Wt, bias, Y, s1, s2, KX, Kpad, D, HW, Btot);
  else if (KX == Kpad)
    mix_gemm_kernel<true, false><<<blocks, 256, 0, stream>>>(
        X, Wt, bias, Y, s1, s2, KX, Kpad, D, HW, Btot);
  else
    mix_gemm_kernel<false, false><<<blocks, 256, 0, stream>>>(
        X, Wt, bias, Y, s1, s2, KX, Kpad, D, HW, Btot);
}

extern "C" void kernel_launch(void* const* d_in, const int* in_sizes, int n_in,
                              void* d_out, int out_size, void* d_ws,
                              size_t ws_size, hipStream_t stream) {
  (void)in_sizes; (void)n_in; (void)out_size; (void)ws_size;
  const float* input = (const float*)d_in[0];
  const int*   label = (const int*)d_in[1];
  const float* emb   = (const float*)d_in[2];
  const float* lw[5] = {(const float*)d_in[3], (const float*)d_in[5],
                        (const float*)d_in[7], (const float*)d_in[9],
                        (const float*)d_in[11]};
  const float* lb[5] = {(const float*)d_in[4], (const float*)d_in[6],
                        (const float*)d_in[8], (const float*)d_in[10],
                        (const float*)d_in[12]};
  const float* cmw[5] = {(const float*)d_in[13], (const float*)d_in[15],
                         (const float*)d_in[17], (const float*)d_in[19],
                         (const float*)d_in[21]};
  const float* cmb[5] = {(const float*)d_in[14], (const float*)d_in[16],
                         (const float*)d_in[18], (const float*)d_in[20],
                         (const float*)d_in[22]};
  const float* bng[4] = {(const float*)d_in[23], (const float*)d_in[25],
                         (const float*)d_in[27], (const float*)d_in[29]};
  const float* bnb[4] = {(const float*)d_in[24], (const float*)d_in[26],
                         (const float*)d_in[28], (const float*)d_in[30]};
  float* out = (float*)d_out;

  // -------- per-layer GEMM config --------
  const int mixK[5]    = {3, 64, 128, 256, 512};    // real weight K
  const int mixKpad[5] = {32, 64, 128, 256, 512};   // padded to KT
  const int mixD[5]    = {64, 128, 256, 512, 512};

  // -------- workspace carve-up --------
  float* ws = (float*)d_ws;
  size_t off = 0;
  float* s1 = ws;           // 512
  float* s2 = ws + 512;     // 512
  off = 1024;
  const int kC[5]  = {3, 64, 128, 256, 512};
  const int kSQ[5] = {16, 16, 9, 16, 4};
  float* dk[5];
  for (int i = 0; i < 5; ++i) { dk[i] = ws + off; off += 128L * kC[i] * kSQ[i]; }
  __bf16* wt[5];
  for (int i = 0; i < 5; ++i) {
    wt[i] = (__bf16*)(ws + off);
    off += (size_t)mixD[i] * mixKpad[i] / 2;        // bf16 elems -> f32 slots
  }
  __bf16* bufA = (__bf16*)(ws + off); off += 8388608 / 2;   // 8.39M bf16
  __bf16* bufB = (__bf16*)(ws + off); off += 33554432 / 2;  // 33.6M bf16

  // -------- weight prep (transpose + bf16, zero-padded) --------
  for (int i = 0; i < 5; ++i) {
    int total = mixD[i] * mixKpad[i];
    wprep_kernel<<<ceil_div(total, 256), 256, 0, stream>>>(
        cmw[i], wt[i], mixK[i], mixKpad[i], mixD[i], total);
  }

  // -------- dynamic kernels, (b, khw, c) layout --------
  for (int i = 0; i < 5; ++i) {
    long total = 128L * kC[i] * kSQ[i];
    dynk_kernel<<<ceil_div(total, 256), 256, 0, stream>>>(
        label, emb, lw[i], lb[i], dk[i], kC[i], kSQ[i], (int)total);
  }

  // -------- layer 1: f32 NCHW input -> bf16 NHWC (8-ch padded) --------
  {
    long total = 128L * 64 * 64 * 8;
    dwconv1_kernel<<<ceil_div(total, 256), 256, 0, stream>>>(
        input, dk[0], bufA, (int)total);
    zero_kernel<<<4, 256, 0, stream>>>(s1, 1024);
    launch_mix(bufA, wt[0], cmb[0], bufB, s1, s2, /*KX=*/8, /*Kpad=*/32,
               /*D=*/64, /*HW=*/4096, 0, stream);
    long t2 = 128L * 4096 * 64;
    bn_bf16_kernel<<<ceil_div(t2 / 8, 256), 256, 0, stream>>>(
        bufB, s1, s2, bng[0], bnb[0], 64, (int)(t2 / 8), 1.f / (128.f * 4096.f));
  }

  // -------- layers 2..5 --------
  const int Cin[4]  = {64, 128, 256, 512};
  const int Cout[4] = {128, 256, 512, 512};
  const int ks[4]   = {4, 3, 4, 2};
  const int st[4]   = {2, 2, 2, 1};
  const int pd[4]   = {1, 1, 1, 0};
  const int Hin[4]  = {64, 32, 16, 8};
  const int Hout[4] = {32, 16, 8, 7};

  for (int L = 0; L < 4; ++L) {
    const int HWo = Hout[L] * Hout[L];
    long ct = 128L * Cin[L] * HWo;
    dwconv_nhwc_kernel<<<ceil_div(ct / 8, 256), 256, 0, stream>>>(
        bufB, dk[L + 1], bufA, Cin[L], Hin[L], Hout[L], ks[L], st[L], pd[L],
        (int)(ct / 8));
    if (L < 3) {
      zero_kernel<<<4, 256, 0, stream>>>(s1, 1024);
      launch_mix(bufA, wt[L + 1], cmb[L + 1], bufB, s1, s2, Cin[L], Cin[L],
                 Cout[L], HWo, 0, stream);
      long t2 = 128L * (long)Cout[L] * HWo;
      bn_bf16_kernel<<<ceil_div(t2 / 8, 256), 256, 0, stream>>>(
          bufB, s1, s2, bng[L + 1], bnb[L + 1], Cout[L], (int)(t2 / 8),
          1.f / (float)(128L * HWo));
    } else {
      // final layer: sigmoid, f32 NCHW straight to d_out
      launch_mix(bufA, wt[4], cmb[4], out, s1, s2, 512, 512, 512, 49, 1,
                 stream);
    }
  }

  // tuple output: append labels after z (128*512*7*7 floats)
  label_copy_kernel<<<1, 128, 0, stream>>>(label, out + 3211264L, 128);
}
